// myGRU_5866925326689
// MI455X (gfx1250) — matrix operand order repaired and checked by reading, not compiled, not run on hardware
//
#include <hip/hip_runtime.h>
#include <hip/hip_bf16.h>
#include <cstdint>

#define SEQ    256
#define BATCH  64
#define VOCAB  128
#define EMBED  256
#define HIDDEN 1024
#define KCAT   (HIDDEN + EMBED)   // 1280 = concatenated [h ; emb] K dimension

#define KH_STEPS   (HIDDEN / 32)  // 32 k-steps in the h region
#define KE_STEPS   (EMBED / 32)   // 8  k-steps in the emb region
#define KH_PER_WV  (KH_STEPS / 4) // 8 per wave
#define KE_PER_WV  (KE_STEPS / 4) // 2 per wave

typedef __attribute__((ext_vector_type(16))) __bf16 v16bf;
typedef __attribute__((ext_vector_type(8)))  float  v8f;

union Frag16 { v16bf v; uint4 q[2]; };

__device__ __forceinline__ unsigned short f2bf(float f) {
    union { float f; uint32_t u; } x; x.f = f;
    uint32_t r = x.u + (0x7FFFu + ((x.u >> 16) & 1u));   // round-to-nearest-even
    return (unsigned short)(r >> 16);
}

// ---------------------------------------------------------------------------
// Prep kernels
// ---------------------------------------------------------------------------

// wcat[g][j][k], k<1024 -> Wh[j][k], else Wx[j][k-1024]; g: 0=r,1=z,2=n
__global__ void k_build_wcat(const float* __restrict__ rx, const float* __restrict__ rh,
                             const float* __restrict__ zx, const float* __restrict__ zh,
                             const float* __restrict__ nx, const float* __restrict__ nh,
                             unsigned short* __restrict__ wcat)
{
    size_t i = (size_t)blockIdx.x * blockDim.x + threadIdx.x;
    if (i >= (size_t)3 * HIDDEN * KCAT) return;
    int g   = (int)(i / ((size_t)HIDDEN * KCAT));
    int rem = (int)(i % ((size_t)HIDDEN * KCAT));
    int j = rem / KCAT, k = rem % KCAT;
    const float* Wh = (g == 0) ? rh : (g == 1) ? zh : nh;
    const float* Wx = (g == 0) ? rx : (g == 1) ? zx : nx;
    float v = (k < HIDDEN) ? Wh[(size_t)j * HIDDEN + k]
                           : Wx[(size_t)j * EMBED + (k - HIDDEN)];
    wcat[i] = f2bf(v);
}

__global__ void k_f32_to_bf16(const float* __restrict__ src,
                              unsigned short* __restrict__ dst, size_t n)
{
    size_t i = (size_t)blockIdx.x * blockDim.x + threadIdx.x;
    if (i < n) dst[i] = f2bf(src[i]);
}

// embA[s][b][e] = bf16(embedding[x[s][b]][e])
__global__ void k_embed(const int* __restrict__ x, const float* __restrict__ emb,
                        unsigned short* __restrict__ embA)
{
    size_t i = (size_t)blockIdx.x * blockDim.x + threadIdx.x;
    if (i >= (size_t)SEQ * BATCH * EMBED) return;
    int e  = (int)(i & (EMBED - 1));
    int sb = (int)(i >> 8);          // EMBED = 256
    int row = x[sb];
    embA[i] = f2bf(emb[(size_t)row * EMBED + e]);
}

__global__ void k_init_h(const float* __restrict__ hprev,
                         float* __restrict__ hf0, unsigned short* __restrict__ hb0,
                         size_t n)
{
    size_t i = (size_t)blockIdx.x * blockDim.x + threadIdx.x;
    if (i < n) { float v = hprev[i]; hf0[i] = v; hb0[i] = f2bf(v); }
}

__global__ void k_copy_hfinal(const float* __restrict__ src, float* __restrict__ dst, size_t n)
{
    size_t i = (size_t)blockIdx.x * blockDim.x + threadIdx.x;
    if (i < n) dst[i] = src[i];
}

// ---------------------------------------------------------------------------
// Recurrent step: one block = one 16(b) x 16(j) tile, 3 gates.
// 4 waves split K: each wave does 8 h-region + 2 emb-region k-steps (uniform,
// compile-time trip counts -> no EXEC divergence around WMMA), LDS reduction.
// ---------------------------------------------------------------------------
__global__ void __launch_bounds__(128)
k_gru_step(const unsigned short* __restrict__ wcat,   // [3][H][KCAT] bf16
           const unsigned short* __restrict__ embA,   // [S][B][E]    bf16
           const unsigned short* __restrict__ hb_cur, // [B][H]       bf16
           unsigned short* __restrict__ hb_next,
           const float* __restrict__ hf_cur,          // [B][H]       f32
           float* __restrict__ hf_next,
           unsigned short* __restrict__ hs,           // [S][B][H]    bf16
           const float* __restrict__ rx_b, const float* __restrict__ rh_b,
           const float* __restrict__ zx_b, const float* __restrict__ zh_b,
           const float* __restrict__ nx_b, const float* __restrict__ nh_b,
           int t)
{
    __shared__ v8f redbuf[4][4][32];                  // [wave][acc][lane]

    const int wave = __builtin_amdgcn_readfirstlane(threadIdx.x >> 5); // scalar wave id
    const int lane = threadIdx.x & 31;
    const int tj = blockIdx.x;                        // H/16 tiles
    const int tb = blockIdx.y;                        // B/16 tiles
    const int hi  = lane >> 4;                        // half-wave id
    const int l16 = lane & 15;

    const int m = tb * 16 + l16;                      // batch row (A matrix M)
    const int n = tj * 16 + l16;                      // hidden col (B matrix N)

    // A-matrix row bases (h region / emb region), per-lane K sub-offset = hi*16
    const unsigned short* arow_h = hb_cur + (size_t)m * HIDDEN + hi * 16;
    const unsigned short* arow_e = embA + ((size_t)t * BATCH + m) * EMBED + hi * 16;

    // B-matrix: Bmat[k][n] = W[n][k]; lane reads 16 contiguous bf16 at n-row
    const size_t boff = (size_t)n * KCAT + (size_t)hi * 16;
    const unsigned short* br = wcat + boff;
    const unsigned short* bz = wcat + (size_t)1 * HIDDEN * KCAT + boff;
    const unsigned short* bn = wcat + (size_t)2 * HIDDEN * KCAT + boff;

    v8f accr = {}, accz = {}, accnh = {}, accnx = {};

    // ---- h region: K = [wave*8*32, wave*8*32 + 8*32) -----------------------
    const int kh0 = wave * KH_PER_WV * 32;
    #pragma unroll
    for (int i = 0; i < KH_PER_WV; ++i) {
        const int k0 = kh0 + i * 32;
        Frag16 a, fr, fz, fn;
        a.q[0]  = *(const uint4*)(arow_h + k0);
        a.q[1]  = *(const uint4*)(arow_h + k0 + 8);
        fr.q[0] = *(const uint4*)(br + k0); fr.q[1] = *(const uint4*)(br + k0 + 8);
        fz.q[0] = *(const uint4*)(bz + k0); fz.q[1] = *(const uint4*)(bz + k0 + 8);
        fn.q[0] = *(const uint4*)(bn + k0); fn.q[1] = *(const uint4*)(bn + k0 + 8);
        accr  = __builtin_amdgcn_wmma_f32_16x16x32_bf16(false, a.v, false, fr.v,
                                                        (short)0, accr,  false, false);
        accz  = __builtin_amdgcn_wmma_f32_16x16x32_bf16(false, a.v, false, fz.v,
                                                        (short)0, accz,  false, false);
        accnh = __builtin_amdgcn_wmma_f32_16x16x32_bf16(false, a.v, false, fn.v,
                                                        (short)0, accnh, false, false);
    }

    // ---- emb region: K = 1024 + [wave*2*32, wave*2*32 + 2*32) --------------
    const int ke0 = wave * KE_PER_WV * 32;
    #pragma unroll
    for (int i = 0; i < KE_PER_WV; ++i) {
        const int ke = ke0 + i * 32;          // offset within emb row
        const int k0 = HIDDEN + ke;           // offset within wcat row
        Frag16 a, fr, fz, fn;
        a.q[0]  = *(const uint4*)(arow_e + ke);
        a.q[1]  = *(const uint4*)(arow_e + ke + 8);
        fr.q[0] = *(const uint4*)(br + k0); fr.q[1] = *(const uint4*)(br + k0 + 8);
        fz.q[0] = *(const uint4*)(bz + k0); fz.q[1] = *(const uint4*)(bz + k0 + 8);
        fn.q[0] = *(const uint4*)(bn + k0); fn.q[1] = *(const uint4*)(bn + k0 + 8);
        accr  = __builtin_amdgcn_wmma_f32_16x16x32_bf16(false, a.v, false, fr.v,
                                                        (short)0, accr,  false, false);
        accz  = __builtin_amdgcn_wmma_f32_16x16x32_bf16(false, a.v, false, fz.v,
                                                        (short)0, accz,  false, false);
        accnx = __builtin_amdgcn_wmma_f32_16x16x32_bf16(false, a.v, false, fn.v,
                                                        (short)0, accnx, false, false);
    }

    redbuf[wave][0][lane] = accr;
    redbuf[wave][1][lane] = accz;
    redbuf[wave][2][lane] = accnh;
    redbuf[wave][3][lane] = accnx;
    __syncthreads();

    if (wave == 0) {
        #pragma unroll
        for (int w = 1; w < 4; ++w) {
            accr  += redbuf[w][0][lane];
            accz  += redbuf[w][1][lane];
            accnh += redbuf[w][2][lane];
            accnx += redbuf[w][3][lane];
        }
        const float biasr = rx_b[n] + rh_b[n];
        const float biasz = zx_b[n] + zh_b[n];
        const float bnhv  = nh_b[n];
        const float bnxv  = nx_b[n];
        #pragma unroll
        for (int v = 0; v < 8; ++v) {
            const int brow = tb * 16 + v + 8 * hi;    // D layout: M = v + 8*(lane/16)
            const size_t idx = (size_t)brow * HIDDEN + n;
            float r  = 1.f / (1.f + __expf(-(accr[v] + biasr)));
            float z  = 1.f / (1.f + __expf(-(accz[v] + biasz)));
            float nn = tanhf(accnx[v] + bnxv + r * (accnh[v] + bnhv));
            float hnew = (1.f - z) * hf_cur[idx] + z * nn;
            hf_next[idx] = hnew;
            unsigned short hb = f2bf(hnew);
            hb_next[idx] = hb;
            hs[(size_t)t * BATCH * HIDDEN + idx] = hb;
        }
    }
}

// ---------------------------------------------------------------------------
// Output projection: logits[m][v] = hs[m][:] . dense_w[v][:] + dense_b[v]
// One wave per 16x16 tile; M=16384, N=128, K=1024.
// ---------------------------------------------------------------------------
__global__ void __launch_bounds__(32)
k_dense(const unsigned short* __restrict__ hs,    // [M][K] bf16
        const unsigned short* __restrict__ dwb,   // [V][K] bf16
        const float* __restrict__ dense_b,
        float* __restrict__ out)                  // [M][V] f32
{
    const int lane = threadIdx.x;
    const int tm = blockIdx.x;                    // M/16
    const int tn = blockIdx.y;                    // V/16
    const int hi = lane >> 4, l16 = lane & 15;

    const unsigned short* arow = hs  + ((size_t)tm * 16 + l16) * HIDDEN + (size_t)hi * 16;
    const unsigned short* brow = dwb + ((size_t)tn * 16 + l16) * HIDDEN + (size_t)hi * 16;

    v8f acc = {};
    #pragma unroll 4
    for (int k0 = 0; k0 < HIDDEN; k0 += 32) {
        Frag16 a, b;
        a.q[0] = *(const uint4*)(arow + k0);
        a.q[1] = *(const uint4*)(arow + k0 + 8);
        b.q[0] = *(const uint4*)(brow + k0);
        b.q[1] = *(const uint4*)(brow + k0 + 8);
        acc = __builtin_amdgcn_wmma_f32_16x16x32_bf16(false, a.v, false, b.v,
                                                      (short)0, acc, false, false);
    }
    const int ncol = tn * 16 + l16;
    const float bias = dense_b[ncol];
    #pragma unroll
    for (int v = 0; v < 8; ++v) {
        const int mrow = tm * 16 + v + 8 * hi;
        out[(size_t)mrow * VOCAB + ncol] = acc[v] + bias;
    }
}

// ---------------------------------------------------------------------------
// Host launcher
// ---------------------------------------------------------------------------
extern "C" void kernel_launch(void* const* d_in, const int* in_sizes, int n_in,
                              void* d_out, int out_size, void* d_ws, size_t ws_size,
                              hipStream_t stream)
{
    (void)in_sizes; (void)n_in; (void)out_size; (void)ws_size;
    const int*   x     = (const int*)  d_in[0];
    const float* hprev = (const float*)d_in[1];
    const float* emb   = (const float*)d_in[2];
    const float* rx_w  = (const float*)d_in[3];
    const float* rx_b  = (const float*)d_in[4];
    const float* rh_w  = (const float*)d_in[5];
    const float* rh_b  = (const float*)d_in[6];
    const float* zx_w  = (const float*)d_in[7];
    const float* zx_b  = (const float*)d_in[8];
    const float* zh_w  = (const float*)d_in[9];
    const float* zh_b  = (const float*)d_in[10];
    const float* nx_w  = (const float*)d_in[11];
    const float* nx_b  = (const float*)d_in[12];
    const float* nh_w  = (const float*)d_in[13];
    const float* nh_b  = (const float*)d_in[14];
    const float* dw    = (const float*)d_in[15];
    const float* db    = (const float*)d_in[16];
    float* out = (float*)d_out;

    uint8_t* ws = (uint8_t*)d_ws;
    size_t off = 0;
    auto take = [&](size_t bytes) -> void* {
        void* p = ws + off;
        off += (bytes + 255) & ~(size_t)255;
        return p;
    };
    unsigned short* wcat = (unsigned short*)take((size_t)3 * HIDDEN * KCAT * 2);  // 7.5 MB
    unsigned short* dwb  = (unsigned short*)take((size_t)VOCAB * HIDDEN * 2);     // 256 KB
    unsigned short* embA = (unsigned short*)take((size_t)SEQ * BATCH * EMBED * 2);// 8 MB
    unsigned short* hb0  = (unsigned short*)take((size_t)BATCH * HIDDEN * 2);
    unsigned short* hb1  = (unsigned short*)take((size_t)BATCH * HIDDEN * 2);
    float*          hf0  = (float*)take((size_t)BATCH * HIDDEN * 4);
    float*          hf1  = (float*)take((size_t)BATCH * HIDDEN * 4);
    unsigned short* hs   = (unsigned short*)take((size_t)SEQ * BATCH * HIDDEN * 2); // 32 MB

    { size_t nel = (size_t)3 * HIDDEN * KCAT;
      k_build_wcat<<<dim3((unsigned)((nel + 255) / 256)), 256, 0, stream>>>(
          rx_w, rh_w, zx_w, zh_w, nx_w, nh_w, wcat); }
    { size_t nel = (size_t)VOCAB * HIDDEN;
      k_f32_to_bf16<<<dim3((unsigned)((nel + 255) / 256)), 256, 0, stream>>>(dw, dwb, nel); }
    { size_t nel = (size_t)SEQ * BATCH * EMBED;
      k_embed<<<dim3((unsigned)((nel + 255) / 256)), 256, 0, stream>>>(x, emb, embA); }
    { size_t nel = (size_t)BATCH * HIDDEN;
      k_init_h<<<dim3((unsigned)((nel + 255) / 256)), 256, 0, stream>>>(hprev, hf0, hb0, nel); }

    for (int t = 0; t < SEQ; ++t) {
        const unsigned short* hbc = (t & 1) ? hb1 : hb0;
        unsigned short*       hbn = (t & 1) ? hb0 : hb1;
        const float*          hfc = (t & 1) ? hf1 : hf0;
        float*                hfn = (t & 1) ? hf0 : hf1;
        k_gru_step<<<dim3(HIDDEN / 16, BATCH / 16), 128, 0, stream>>>(
            wcat, embA, hbc, hbn, hfc, hfn, hs,
            rx_b, rh_b, zx_b, zh_b, nx_b, nh_b, t);
    }

    k_dense<<<dim3(SEQ * BATCH / 16, VOCAB / 16), 32, 0, stream>>>(hs, dwb, db, out);

    { size_t nel = (size_t)BATCH * HIDDEN;   // after 256 steps, final h lives in buffer 0
      k_copy_hfinal<<<dim3((unsigned)((nel + 255) / 256)), 256, 0, stream>>>(
          hf0, out + (size_t)SEQ * BATCH * VOCAB, nel); }
}
